// Block_46411416600785
// MI455X (gfx1250) — matrix operand order repaired and checked by reading, not compiled, not run on hardware
//
#include <hip/hip_runtime.h>
#include <hip/hip_bf16.h>
#include <cstdint>

typedef __attribute__((ext_vector_type(16))) _Float16 v16h;
typedef __attribute__((ext_vector_type(8)))  _Float16 v8h;
typedef __attribute__((ext_vector_type(8)))  float    v8f;
typedef __attribute__((ext_vector_type(4)))  int      v4i;

#define WMMA_F32_F16(a, b, c) \
  __builtin_amdgcn_wmma_f32_16x16x32_f16(false, (a), false, (b), (short)0, (c), false, false)

// ---------------------------------------------------------------------------
// CDNA5 async global->LDS copy (GLOBAL_LOAD_ASYNC_TO_LDS_B128, ASYNCcnt).
// Signature probed: (v4i* global_src, AS3 v4i* lds_dst, imm, imm) — bound via
// an expression-SFINAE ladder so alternative toolchains still compile.
#if defined(__HIP_DEVICE_COMPILE__) &&                                  \
    __has_builtin(__builtin_amdgcn_global_load_async_to_lds_b128) &&    \
    __has_builtin(__builtin_amdgcn_s_wait_asynccnt)
#define USE_ASYNC_LDS 1
#else
#define USE_ASYNC_LDS 0
#endif

#if USE_ASYNC_LDS
typedef __attribute__((address_space(3))) v4i as3_v4i;
typedef as3_v4i* v4i_lds;

struct prio0 {};
struct prio1 : prio0 {};
struct prio2 : prio1 {};
struct prio3 : prio2 {};

template <typename G, typename L3>
__device__ __forceinline__ auto async_impl(G g, L3 l3, void* lgen, prio3)
    -> decltype(__builtin_amdgcn_global_load_async_to_lds_b128(g, l3, 0, 0)) {
  __builtin_amdgcn_global_load_async_to_lds_b128(g, l3, 0, 0);
}
template <typename G, typename L3>
__device__ __forceinline__ auto async_impl(G g, L3 l3, void* lgen, prio2)
    -> decltype(__builtin_amdgcn_global_load_async_to_lds_b128(g, (v4i*)lgen, 0, 0)) {
  __builtin_amdgcn_global_load_async_to_lds_b128(g, (v4i*)lgen, 0, 0);
}
template <typename G, typename L3>
__device__ __forceinline__ auto async_impl(G g, L3 l3, void* lgen, prio1)
    -> decltype(__builtin_amdgcn_global_load_async_to_lds_b128(g, l3, 0)) {
  __builtin_amdgcn_global_load_async_to_lds_b128(g, l3, 0);
}
template <typename G, typename L3>
__device__ __forceinline__ void async_impl(G g, L3, void* lgen, prio0) {
  *(v4i*)lgen = *g;                      // synchronous fallback
}

__device__ __forceinline__ void async_b128(const void* g, void* l) {
  v4i*    gf = (v4i*)g;                                   // generic/flat src
  v4i_lds l3 = (v4i_lds)(uint32_t)(uintptr_t)l;           // LDS (AS3) dst
  async_impl(gf, l3, l, prio3{});
}
#define ASYNC_COMMIT() __builtin_amdgcn_s_wait_asynccnt(0)
#else
#define ASYNC_COMMIT() ((void)0)
#endif

// ---------------------------------------------------------------------------
// Fragment gathers from LDS (wave32, v_wmma_f32_16x16x32_f16 layouts).
// A 16x32 f16: lane<16 -> row=lane, K {0..7, 16..23}; lane>=16 -> row=lane-16,
//              K {8..15, 24..31}.  p = &lds[row*ld + kstep + (lane>>4)*8]
__device__ __forceinline__ v16h frag_a(const _Float16* p) {
  v8h lo = *(const v8h*)p;           // K +0..7
  v8h hi = *(const v8h*)(p + 16);    // K +16..23
  v16h r;
#pragma unroll
  for (int i = 0; i < 8; ++i) { r[i] = lo[i]; r[i + 8] = hi[i]; }
  return r;
}
// B 32x16 f16 (LDS holds B transposed, [n][k]): lane<16 -> col=lane, K {0..7, 16..23};
// lane>=16 -> col=lane-16, K {8..15, 24..31}.  p = &lds[n*ld + kstep + (lane>>4)*8]
__device__ __forceinline__ v16h frag_b(const _Float16* p) {
  v8h lo = *(const v8h*)p;           // K +0..7
  v8h hi = *(const v8h*)(p + 16);    // K +16..23
  v16h r;
#pragma unroll
  for (int i = 0; i < 8; ++i) { r[i] = lo[i]; r[i + 8] = hi[i]; }
  return r;
}

// ---------------------------------------------------------------------------
// Weight conversion.  All GEMM B-operands are weights, so they are emitted
// PRE-TRANSPOSED ([N][K] f16): the GEMM's LDS image is [n][k], so tile copies
// become contiguous 16B chunks eligible for the async global->LDS engine.
// Wq/Wk/Wv [H,C,HS] f32 -> Wqkv^T f16 [3072][1024]; n = sel*1024 + h*64 + d.
__global__ __launch_bounds__(256) void cvt_qkv_w(
    const float* __restrict__ Wq, const float* __restrict__ Wk,
    const float* __restrict__ Wv, _Float16* __restrict__ out) {
  int i = blockIdx.x * 256 + threadIdx.x;          // over 3072*1024, k fastest
  int n = i >> 10, k = i & 1023;
  int sel = n >> 10, hh = (n & 1023) >> 6, d = n & 63;
  const float* W = (sel == 0) ? Wq : (sel == 1) ? Wk : Wv;
  out[i] = (_Float16)W[((size_t)hh * 1024 + k) * 64 + d];
}

// out[n*K + k] = (f16) in[k*N + n]   (K = 1<<kbits)
__global__ __launch_bounds__(256) void cvt_f32_f16_t(
    const float* __restrict__ in, _Float16* __restrict__ out, int kbits, int N) {
  int i = blockIdx.x * 256 + threadIdx.x;
  int n = i >> kbits, k = i & ((1 << kbits) - 1);
  out[i] = (_Float16)in[(size_t)k * N + n];
}

// ---------------------------------------------------------------------------
// LayerNorm: one 256-thread block per row of 1024, f32 in -> f16 out.
__global__ __launch_bounds__(256) void ln_kernel(
    const float* __restrict__ x, const float* __restrict__ g,
    const float* __restrict__ bb, _Float16* __restrict__ out) {
  const int row = blockIdx.x;
  const int tid = threadIdx.x;
  const float* xr = x + (size_t)row * 1024;
  float v[4], s = 0.f, ss = 0.f;
#pragma unroll
  for (int j = 0; j < 4; ++j) {
    float t = xr[tid + j * 256];
    v[j] = t; s += t; ss += t * t;
  }
#pragma unroll
  for (int off = 16; off >= 1; off >>= 1) {
    s  += __shfl_xor(s,  off, 32);
    ss += __shfl_xor(ss, off, 32);
  }
  __shared__ float red[16];
  int wid = tid >> 5, lane = tid & 31;
  if (lane == 0) { red[wid] = s; red[wid + 8] = ss; }
  __syncthreads();
  if (tid == 0) {
    float a = 0.f, b2 = 0.f;
#pragma unroll
    for (int i = 0; i < 8; ++i) { a += red[i]; b2 += red[i + 8]; }
    red[0] = a; red[8] = b2;
  }
  __syncthreads();
  float mean = red[0] * (1.0f / 1024.0f);
  float var  = red[8] * (1.0f / 1024.0f) - mean * mean;
  float rstd = rsqrtf(var + 1e-5f);
  _Float16* orow = out + (size_t)row * 1024;
#pragma unroll
  for (int j = 0; j < 4; ++j) {
    int c = tid + j * 256;
    orow[c] = (_Float16)((v[j] - mean) * rstd * g[c] + bb[c]);
  }
}

// ---------------------------------------------------------------------------
// WMMA GEMM: out[M,N] = A[M,K](f16, row-major) * B (f16, PRE-TRANSPOSED [N][K])
// [+bias][+resid][ReLU].  128x128 macro-tile, 8 waves (4x2), 32x64 per wave,
// K-step 32.  Ping-pong LDS double buffering; BOTH tiles stream global->LDS
// via GLOBAL_LOAD_ASYNC_TO_LDS_B128 (no staging registers, no LDS scatter).
template <bool BIAS, bool RELU, bool RESID, bool OUTF16>
__global__ __launch_bounds__(256) void gemm_wmma(
    const _Float16* __restrict__ A, const _Float16* __restrict__ Bt,
    const float* __restrict__ bias, const float* __restrict__ resid,
    _Float16* __restrict__ outh, float* __restrict__ outf, int N, int K) {
  __shared__ _Float16 As[2][128 * 40];   // [m][k], ld=40 (80B rows, 16B aligned)
  __shared__ _Float16 Bs[2][128 * 40];   // [n][k]
  const int tid  = threadIdx.x;
  const int lane = tid & 31;
  const int wid  = tid >> 5;
  const int wm = wid & 3, wn = wid >> 2;
  const int m0 = blockIdx.y * 128, n0 = blockIdx.x * 128;
  const int l15  = lane & 15;
  const int hi8  = (lane >> 4) << 3;
  const int hi16 = (lane >> 4) << 4;

  // per-thread copy coordinates
  const int ar = tid >> 2;               // A rows ar, ar+64 ; cols ac..ac+7
  const int ac = (tid & 3) << 3;
  const int br = tid >> 1;               // B n-row 0..127 ; k-chunks bkc, bkc+8
  const int bkc = (tid & 1) << 4;

  const _Float16* Ag0 = A  + (size_t)(m0 + ar) * K + ac;
  const _Float16* Ag1 = A  + (size_t)(m0 + ar + 64) * K + ac;
  const _Float16* Bg  = Bt + (size_t)(n0 + br) * K + bkc;

  // ---- prologue: tile 0 -> buffer 0
  {
#if USE_ASYNC_LDS
    async_b128(Ag0,    &As[0][ar * 40 + ac]);
    async_b128(Ag1,    &As[0][(ar + 64) * 40 + ac]);
    async_b128(Bg,     &Bs[0][br * 40 + bkc]);
    async_b128(Bg + 8, &Bs[0][br * 40 + bkc + 8]);
#else
    *(v8h*)&As[0][ar * 40 + ac]        = *(const v8h*)Ag0;
    *(v8h*)&As[0][(ar + 64) * 40 + ac] = *(const v8h*)Ag1;
    *(v8h*)&Bs[0][br * 40 + bkc]       = *(const v8h*)Bg;
    *(v8h*)&Bs[0][br * 40 + bkc + 8]   = *(const v8h*)(Bg + 8);
#endif
    ASYNC_COMMIT();
    __syncthreads();
  }

  v8f acc[2][4] = {};
  const int KT = K >> 5;
  for (int it = 0; it < KT; ++it) {
    const int cur = it & 1, nxt = cur ^ 1;
    const int koff = (it + 1) << 5;
    const bool pf = (it + 1 < KT);
#if USE_ASYNC_LDS
    if (pf) {                            // fire-and-forget next tile
      async_b128(Ag0 + koff,    &As[nxt][ar * 40 + ac]);
      async_b128(Ag1 + koff,    &As[nxt][(ar + 64) * 40 + ac]);
      async_b128(Bg + koff,     &Bs[nxt][br * 40 + bkc]);
      async_b128(Bg + koff + 8, &Bs[nxt][br * 40 + bkc + 8]);
    }
#else
    v8h a0 = {}, a1 = {}, b0 = {}, b1 = {};
    if (pf) {                            // stage next tile in registers
      a0 = *(const v8h*)(Ag0 + koff);
      a1 = *(const v8h*)(Ag1 + koff);
      b0 = *(const v8h*)(Bg + koff);
      b1 = *(const v8h*)(Bg + koff + 8);
    }
#endif

    // compute on buffer `cur` (overlaps the in-flight copies)
    v16h af[2], bf4[4];
#pragma unroll
    for (int mi = 0; mi < 2; ++mi)
      af[mi] = frag_a(&As[cur][(wm * 32 + mi * 16 + l15) * 40 + hi8]);
#pragma unroll
    for (int ni = 0; ni < 4; ++ni)
      bf4[ni] = frag_b(&Bs[cur][(wn * 64 + ni * 16 + l15) * 40 + hi8]);
#pragma unroll
    for (int mi = 0; mi < 2; ++mi)
#pragma unroll
      for (int ni = 0; ni < 4; ++ni)
        acc[mi][ni] = WMMA_F32_F16(af[mi], bf4[ni], acc[mi][ni]);

#if !USE_ASYNC_LDS
    if (pf) {                            // commit staged tile
      *(v8h*)&As[nxt][ar * 40 + ac]        = a0;
      *(v8h*)&As[nxt][(ar + 64) * 40 + ac] = a1;
      *(v8h*)&Bs[nxt][br * 40 + bkc]       = b0;
      *(v8h*)&Bs[nxt][br * 40 + bkc + 8]   = b1;
    }
#endif
    ASYNC_COMMIT();
    __syncthreads();
  }

  // epilogue
#pragma unroll
  for (int mi = 0; mi < 2; ++mi)
#pragma unroll
    for (int ni = 0; ni < 4; ++ni)
#pragma unroll
      for (int r = 0; r < 8; ++r) {
        int gm = m0 + wm * 32 + mi * 16 + r + hi8;
        int gn = n0 + wn * 64 + ni * 16 + l15;
        float v = acc[mi][ni][r];
        if (BIAS)  v += bias[gn];
        if (RESID) v += resid[(size_t)gm * N + gn];
        if (RELU)  v = fmaxf(v, 0.0f);
        if (OUTF16) outh[(size_t)gm * N + gn] = (_Float16)v;
        else        outf[(size_t)gm * N + gn] = v;
      }
}

// ---------------------------------------------------------------------------
// Block-causal flash attention.  Staircase mask == "attend to all keys
// s < (floor(t/64)+1)*64", so 64-aligned query tiles need NO masking.
// Grid (qb=16, h=16, b=8), 128 threads = 4 waves; wave w owns Q rows 16w..16w+15.
// K/V tiles are ping-pong double buffered; K-tile uses the async copy path.
__global__ __launch_bounds__(128) void attn_kernel(
    const _Float16* __restrict__ qkv,   // [8192, 3072]; q @0, k @1024, v @2048
    _Float16* __restrict__ outp) {      // [8192, 1024] layout [b,t,h*64+d]
  const int qb = blockIdx.x, h = blockIdx.y, b = blockIdx.z;
  const int tid = threadIdx.x, lane = tid & 31, w = tid >> 5;
  const int l15 = lane & 15;
  const int hi8  = (lane >> 4) << 3;
  const int hi16 = (lane >> 4) << 4;

  __shared__ _Float16 Qs[64 * 72];        // [t][d]
  __shared__ _Float16 Ks[2][64 * 72];     // [s][d]  (== B^T for Q*K^T)
  __shared__ _Float16 Vs[2][64 * 72];     // [d][s]  (== B^T for P*V)
  __shared__ _Float16 Ps[4][16 * 72];     // per-wave P tile [t][s]

  const int rowbase = b * 1024;
  const int colq = h * 64;
  // per-thread copy coordinates: rows tr+16j (j=0..3), cols tc..tc+7
  const int tr = tid >> 3;
  const int tc = (tid & 7) << 3;

  // ---- prologue: Q tile + (K,V) block 0 -> buffer 0
#pragma unroll
  for (int j = 0; j < 4; ++j) {
    const int r = tr + 16 * j;
    const size_t src = (size_t)(rowbase + qb * 64 + r) * 3072 + colq;
    *(v8h*)&Qs[r * 72 + tc] = *(const v8h*)&qkv[src + tc];
  }
#pragma unroll
  for (int j = 0; j < 4; ++j) {
    const int r = tr + 16 * j;
    const size_t src = (size_t)(rowbase + r) * 3072 + colq;
#if USE_ASYNC_LDS
    async_b128(&qkv[src + 1024 + tc], &Ks[0][r * 72 + tc]);
#else
    *(v8h*)&Ks[0][r * 72 + tc] = *(const v8h*)&qkv[src + 1024 + tc];
#endif
    v8h vv = *(const v8h*)&qkv[src + 2048 + tc];
#pragma unroll
    for (int e = 0; e < 8; ++e) Vs[0][(tc + e) * 72 + r] = vv[e];  // transpose
  }
  ASYNC_COMMIT();
  __syncthreads();

  v16h qf[2];
#pragma unroll
  for (int ks = 0; ks < 2; ++ks)
    qf[ks] = frag_a(&Qs[(w * 16 + l15) * 72 + ks * 32 + hi8]);

  v8f o[4] = {};
  float m[8], l[8];
#pragma unroll
  for (int r = 0; r < 8; ++r) { m[r] = -1e30f; l[r] = 0.0f; }

  for (int kb = 0; kb <= qb; ++kb) {
    const int cur = kb & 1, nxt = cur ^ 1;
    const bool pf = (kb < qb);
    v8h vst[4] = {};
#if !USE_ASYNC_LDS
    v8h kst[4] = {};
#endif
    if (pf) {                            // issue next K/V block
#pragma unroll
      for (int j = 0; j < 4; ++j) {
        const int r = tr + 16 * j;
        const size_t src = (size_t)(rowbase + (kb + 1) * 64 + r) * 3072 + colq;
#if USE_ASYNC_LDS
        async_b128(&qkv[src + 1024 + tc], &Ks[nxt][r * 72 + tc]);
#else
        kst[j] = *(const v8h*)&qkv[src + 1024 + tc];
#endif
        vst[j] = *(const v8h*)&qkv[src + 2048 + tc];
      }
    }

    // S = (Q K^T) * HS^-0.5  -> 16x64 per wave
    v8f s[4];
#pragma unroll
    for (int nt = 0; nt < 4; ++nt) {
      v8f z = {};
#pragma unroll
      for (int ks = 0; ks < 2; ++ks) {
        v16h bf = frag_b(&Ks[cur][(nt * 16 + l15) * 72 + ks * 32 + hi8]);
        z = WMMA_F32_F16(qf[ks], bf, z);
      }
#pragma unroll
      for (int r = 0; r < 8; ++r) z[r] *= 0.125f;   // 1/sqrt(64)
      s[nt] = z;
    }

    // online softmax (rows live in 16-lane halves)
#pragma unroll
    for (int r = 0; r < 8; ++r) {
      float mx = -1e30f;
#pragma unroll
      for (int nt = 0; nt < 4; ++nt) mx = fmaxf(mx, s[nt][r]);
#pragma unroll
      for (int off = 1; off < 16; off <<= 1) mx = fmaxf(mx, __shfl_xor(mx, off, 32));
      float mnew = fmaxf(m[r], mx);
      float corr = __expf(m[r] - mnew);
      m[r] = mnew;
      float rs = 0.0f;
#pragma unroll
      for (int nt = 0; nt < 4; ++nt) {
        float p = __expf(s[nt][r] - mnew);
        s[nt][r] = p; rs += p;
      }
#pragma unroll
      for (int off = 1; off < 16; off <<= 1) rs += __shfl_xor(rs, off, 32);
      l[r] = l[r] * corr + rs;
#pragma unroll
      for (int nt = 0; nt < 4; ++nt) o[nt][r] *= corr;
    }

    // P (f32 C-layout) -> f16 A-layout via per-wave LDS round-trip
#pragma unroll
    for (int nt = 0; nt < 4; ++nt)
#pragma unroll
      for (int r = 0; r < 8; ++r)
        Ps[w][(r + hi8) * 72 + nt * 16 + l15] = (_Float16)s[nt][r];

    // O += P * V
#pragma unroll
    for (int ks = 0; ks < 2; ++ks) {
      v16h pfr = frag_a(&Ps[w][l15 * 72 + ks * 32 + hi8]);
#pragma unroll
      for (int nt = 0; nt < 4; ++nt) {
        v16h vf = frag_b(&Vs[cur][(nt * 16 + l15) * 72 + ks * 32 + hi8]);
        o[nt] = WMMA_F32_F16(pfr, vf, o[nt]);
      }
    }

    if (pf) {                            // commit staged V (transposed) [+K]
#pragma unroll
      for (int j = 0; j < 4; ++j) {
        const int r = tr + 16 * j;
#pragma unroll
        for (int e = 0; e < 8; ++e) Vs[nxt][(tc + e) * 72 + r] = vst[j][e];
#if !USE_ASYNC_LDS
        *(v8h*)&Ks[nxt][r * 72 + tc] = kst[j];
#endif
      }
    }
    ASYNC_COMMIT();
    __syncthreads();
  }

  // epilogue: O / l, write f16 [b, t, h*64+d]
#pragma unroll
  for (int nt = 0; nt < 4; ++nt)
#pragma unroll
    for (int r = 0; r < 8; ++r) {
      int grow = rowbase + qb * 64 + w * 16 + r + hi8;
      int gcol = colq + nt * 16 + l15;
      outp[(size_t)grow * 1024 + gcol] = (_Float16)(o[nt][r] / l[r]);
    }
}

// ---------------------------------------------------------------------------
extern "C" void kernel_launch(void* const* d_in, const int* in_sizes, int n_in,
                              void* d_out, int out_size, void* d_ws, size_t ws_size,
                              hipStream_t stream) {
  (void)in_sizes; (void)n_in; (void)out_size; (void)ws_size;
  const float* x    = (const float*)d_in[0];
  // d_in[1] = mask: staircase structure exploited analytically, not read.
  const float* Wq   = (const float*)d_in[2];
  const float* Wk   = (const float*)d_in[3];
  const float* Wv   = (const float*)d_in[4];
  const float* Wo   = (const float*)d_in[5];
  const float* bo   = (const float*)d_in[6];
  const float* ln1g = (const float*)d_in[7];
  const float* ln1b = (const float*)d_in[8];
  const float* ln2g = (const float*)d_in[9];
  const float* ln2b = (const float*)d_in[10];
  const float* W1   = (const float*)d_in[11];
  const float* b1   = (const float*)d_in[12];
  const float* W2   = (const float*)d_in[13];
  const float* b2   = (const float*)d_in[14];
  float* out = (float*)d_out;

  char* ws = (char*)d_ws;
  _Float16* wqkv = (_Float16*)(ws + 0);            //  6 MB [3072][1024] (W^T)
  _Float16* wo   = (_Float16*)(ws + 6291456);      //  2 MB [1024][1024] (W^T)
  _Float16* w1   = (_Float16*)(ws + 8388608);      //  8 MB [4096][1024] (W^T)
  _Float16* w2   = (_Float16*)(ws + 16777216);     //  8 MB [1024][4096] (W^T)
  _Float16* hbuf = (_Float16*)(ws + 25165824);     // 16 MB (h, reused as h2)
  _Float16* qkv  = (_Float16*)(ws + 41943040);     // 48 MB [8192,3072]
  _Float16* att  = (_Float16*)(ws + 92274688);     // 16 MB [8192,1024]
  float*    xmid = (float*)   (ws + 109051904);    // 32 MB [8192,1024]
  _Float16* ff1  = (_Float16*)(ws + 142606336);    // 64 MB [8192,4096]

  // weight conversion (f32 -> f16, transposed to [N][K])
  cvt_qkv_w    <<<dim3(3072 * 1024 / 256), 256, 0, stream>>>(Wq, Wk, Wv, wqkv);
  cvt_f32_f16_t<<<dim3(1024 * 1024 / 256), 256, 0, stream>>>(Wo, wo, 10, 1024);
  cvt_f32_f16_t<<<dim3(4096 * 1024 / 256), 256, 0, stream>>>(W1, w1, 10, 4096);
  cvt_f32_f16_t<<<dim3(1024 * 4096 / 256), 256, 0, stream>>>(W2, w2, 12, 1024);

  // h = LN1(x)
  ln_kernel<<<dim3(8192), 256, 0, stream>>>(x, ln1g, ln1b, hbuf);
  // qkv = h @ Wqkv
  gemm_wmma<false, false, false, true>
      <<<dim3(3072 / 128, 8192 / 128), 256, 0, stream>>>(
          hbuf, wqkv, nullptr, nullptr, qkv, nullptr, 3072, 1024);
  // attention (block-causal, online softmax)
  attn_kernel<<<dim3(16, 16, 8), 128, 0, stream>>>(qkv, att);
  // xmid = x + att @ Wo + bo
  gemm_wmma<true, false, true, false>
      <<<dim3(1024 / 128, 8192 / 128), 256, 0, stream>>>(
          att, wo, bo, x, nullptr, xmid, 1024, 1024);
  // h2 = LN2(xmid)
  ln_kernel<<<dim3(8192), 256, 0, stream>>>(xmid, ln2g, ln2b, hbuf);
  // ff1 = relu(h2 @ W1 + b1)
  gemm_wmma<true, true, false, true>
      <<<dim3(4096 / 128, 8192 / 128), 256, 0, stream>>>(
          hbuf, w1, b1, nullptr, ff1, nullptr, 4096, 1024);
  // out = xmid + ff1 @ W2 + b2
  gemm_wmma<true, false, true, false>
      <<<dim3(1024 / 128, 8192 / 128), 256, 0, stream>>>(
          ff1, w2, b2, xmid, nullptr, out, 1024, 4096);
}
